// CosineRouter_61658550501600
// MI455X (gfx1250) — matrix operand order repaired and checked by reading, not compiled, not run on hardware
//
#include <hip/hip_runtime.h>
#include <hip/hip_bf16.h>

// Problem constants (match reference setup_inputs)
#define TOKS   16384
#define DIM    4096
#define NEXP   64
#define NCOL   128        // E * P
#define TOPK   8
#define KCH    256        // K-chunk staged in LDS per iteration
#define TILET  32         // tokens per block (2 x 16-row WMMA tiles)

typedef __attribute__((ext_vector_type(16))) __bf16       v16bf;
typedef __attribute__((ext_vector_type(8)))  float        v8f;
typedef __attribute__((ext_vector_type(4)))  float        v4f;
typedef __attribute__((ext_vector_type(4)))  unsigned int v4u;

union FragU { v4u q[2]; v16bf v; };

__device__ __forceinline__ unsigned short f2bf(float f) {
    unsigned int u = __float_as_uint(f);
    u += 0x7fffu + ((u >> 16) & 1u);          // round-to-nearest-even
    return (unsigned short)(u >> 16);
}

__device__ __forceinline__ float wred_max(float v) {
    #pragma unroll
    for (int o = 16; o > 0; o >>= 1) v = fmaxf(v, __shfl_xor(v, o, 32));
    return v;
}
__device__ __forceinline__ float wred_sum(float v) {
    #pragma unroll
    for (int o = 16; o > 0; o >>= 1) v += __shfl_xor(v, o, 32);
    return v;
}
__device__ __forceinline__ int wred_min_i(int v) {
    #pragma unroll
    for (int o = 16; o > 0; o >>= 1) v = min(v, __shfl_xor(v, o, 32));
    return v;
}

// ---------------------------------------------------------------------------
// Pass 1: per-row inverse norms. Blocks [0, NCOL) handle prototype rows and
// emit pre-normalized bf16 prototypes; blocks [NCOL, NCOL+TOKS) emit token
// inverse norms only (tokens are scaled on the fly during LDS staging).
// ---------------------------------------------------------------------------
__global__ void norm_cast_kernel(const float* __restrict__ h,
                                 const float* __restrict__ protos,
                                 unsigned short* __restrict__ pb,
                                 float* __restrict__ hn) {
    __shared__ float red[8];
    __shared__ float s_scale;
    const int bid  = blockIdx.x;
    const int tid  = threadIdx.x;
    const int lane = tid & 31;
    const int wv   = tid >> 5;

    const bool isProto = (bid < NCOL);
    const float* src = isProto ? (protos + (size_t)bid * DIM)
                               : (h + (size_t)(bid - NCOL) * DIM);

    // 256 threads * 16 contiguous floats = 4096
    v4f r[4];
    float ss = 0.f;
    #pragma unroll
    for (int i = 0; i < 4; ++i) {
        r[i] = *(const v4f*)(src + tid * 16 + i * 4);
        ss += r[i].x * r[i].x + r[i].y * r[i].y + r[i].z * r[i].z + r[i].w * r[i].w;
    }
    ss = wred_sum(ss);
    if (lane == 0) red[wv] = ss;
    __syncthreads();
    if (tid == 0) {
        float t = 0.f;
        #pragma unroll
        for (int i = 0; i < 8; ++i) t += red[i];
        s_scale = 1.0f / (sqrtf(t) + 1e-6f);
    }
    __syncthreads();
    const float sc = s_scale;

    if (isProto) {
        unsigned int pk[8];
        #pragma unroll
        for (int i = 0; i < 4; ++i) {
            pk[2 * i]     = (unsigned int)f2bf(r[i].x * sc) | ((unsigned int)f2bf(r[i].y * sc) << 16);
            pk[2 * i + 1] = (unsigned int)f2bf(r[i].z * sc) | ((unsigned int)f2bf(r[i].w * sc) << 16);
        }
        v4u* dst = (v4u*)(pb + (size_t)bid * DIM + tid * 16);
        v4u a, b;
        a[0] = pk[0]; a[1] = pk[1]; a[2] = pk[2]; a[3] = pk[3];
        b[0] = pk[4]; b[1] = pk[5]; b[2] = pk[6]; b[3] = pk[7];
        dst[0] = a;
        dst[1] = b;
    } else {
        if (tid == 0) hn[bid - NCOL] = sc;
    }
}

// ---------------------------------------------------------------------------
// Pass 2: fused GEMM (bf16 WMMA, f32 acc) + logsumexp + top-8 + masked softmax.
// Block = 256 threads = 8 waves, owns 32 tokens x all 128 proto columns.
// Wave w computes two 16x16 tiles (token rows 0-15 and 16-31) for columns
// [16w, 16w+16) = experts [8w, 8w+8), sharing one B fragment per K-step.
// A tile is double-buffered in LDS (one barrier per K-chunk).
// ---------------------------------------------------------------------------
__global__ void router_kernel(const float* __restrict__ h,
                              const unsigned short* __restrict__ pb,
                              const float* __restrict__ hn,
                              float* __restrict__ out) {
    __shared__ unsigned short lds_a[2][TILET][KCH];   // 2 x 16 KB: scaled bf16 A
    __shared__ float lds_logit[TILET][NEXP];          // 8 KB: pooled logits

    const int tid  = threadIdx.x;
    const int lane = tid & 31;
    const int wv   = tid >> 5;          // 0..7
    const int hlf  = lane >> 4;         // 0/1 half-wave
    const int l15  = lane & 15;
    const int tokBase = blockIdx.x * TILET;

    // staging role: thread -> (row 0..31, 32-float column block)
    const int srow = tid >> 3;
    const int scol = (tid & 7) * 32;
    const float sscale = hn[tokBase + srow];
    const float* srcRow = h + (size_t)(tokBase + srow) * DIM;

    // B fragment base: column = wv*16 + l15; per-lane K-run starts at hlf*16
    const unsigned short* bRow = pb + (size_t)(wv * 16 + l15) * DIM;

    v8f acc0 = {};
    v8f acc1 = {};
    v4f r[8];

    // prologue: fetch + stage chunk 0 into buffer 0
    #pragma unroll
    for (int i = 0; i < 8; ++i) r[i] = *(const v4f*)(srcRow + scol + i * 4);
    {
        unsigned int pk[16];
        #pragma unroll
        for (int i = 0; i < 8; ++i) {
            pk[2 * i]     = (unsigned int)f2bf(r[i].x * sscale) | ((unsigned int)f2bf(r[i].y * sscale) << 16);
            pk[2 * i + 1] = (unsigned int)f2bf(r[i].z * sscale) | ((unsigned int)f2bf(r[i].w * sscale) << 16);
        }
        v4u* dst = (v4u*)&lds_a[0][srow][scol];
        #pragma unroll
        for (int j = 0; j < 4; ++j) {
            v4u q;
            q[0] = pk[4 * j + 0]; q[1] = pk[4 * j + 1];
            q[2] = pk[4 * j + 2]; q[3] = pk[4 * j + 3];
            dst[j] = q;
        }
    }
    __syncthreads();

    const int NCHUNK = DIM / KCH;       // 16
    for (int ci = 0; ci < NCHUNK; ++ci) {
        const int  cur  = ci & 1;
        const bool more = (ci + 1 < NCHUNK);

        // prefetch next fp32 A chunk into registers (overlaps with WMMAs)
        if (more) {
            const float* p = srcRow + (ci + 1) * KCH + scol;
            #pragma unroll
            for (int i = 0; i < 8; ++i) r[i] = *(const v4f*)(p + i * 4);
        }

        // ---- 8 K-steps x 2 token tiles over this chunk ----
        #pragma unroll
        for (int kk = 0; kk < KCH; kk += 32) {
            FragU a0, a1, b;
            // A 16x32 bf16: lane row = l15; contiguous K-run of 16 at base = hlf*16
            const unsigned short* ap0 = &lds_a[cur][l15][kk + hlf * 16];
            const unsigned short* ap1 = &lds_a[cur][16 + l15][kk + hlf * 16];
            a0.q[0] = *(const v4u*)ap0;           // +0  elems (16B)
            a0.q[1] = *(const v4u*)(ap0 + 8);     // +8  elems (16B)
            a1.q[0] = *(const v4u*)ap1;
            a1.q[1] = *(const v4u*)(ap1 + 8);
            // B 32x16 bf16: lane col = l15; contiguous K-run of 16 at hlf*16
            const unsigned short* bp = bRow + ci * KCH + kk + hlf * 16;
            b.q[0] = *(const v4u*)bp;
            b.q[1] = *(const v4u*)(bp + 8);
            acc0 = __builtin_amdgcn_wmma_f32_16x16x32_bf16(
                false, a0.v, false, b.v, (short)0, acc0, false, false);
            acc1 = __builtin_amdgcn_wmma_f32_16x16x32_bf16(
                false, a1.v, false, b.v, (short)0, acc1, false, false);
        }

        // stage next chunk into the other buffer while this one is still live
        if (more) {
            unsigned int pk[16];
            #pragma unroll
            for (int i = 0; i < 8; ++i) {
                pk[2 * i]     = (unsigned int)f2bf(r[i].x * sscale) | ((unsigned int)f2bf(r[i].y * sscale) << 16);
                pk[2 * i + 1] = (unsigned int)f2bf(r[i].z * sscale) | ((unsigned int)f2bf(r[i].w * sscale) << 16);
            }
            v4u* dst = (v4u*)&lds_a[cur ^ 1][srow][scol];
            #pragma unroll
            for (int j = 0; j < 4; ++j) {
                v4u q;
                q[0] = pk[4 * j + 0]; q[1] = pk[4 * j + 1];
                q[2] = pk[4 * j + 2]; q[3] = pk[4 * j + 3];
                dst[j] = q;
            }
        }
        __syncthreads();
    }

    // ---- epilogue 1: logsumexp over the P=2 prototype pair (adjacent lanes) ----
    // C layout: acc[i] @ lane -> M = i + 8*hlf (+16 for tile 1), N = l15
    #pragma unroll
    for (int i = 0; i < 8; ++i) {
        float s0  = acc0[i];
        float s1  = acc1[i];
        float so0 = __shfl_xor(s0, 1, 32);     // partner prototype of same expert
        float so1 = __shfl_xor(s1, 1, 32);
        if ((l15 & 1) == 0) {
            const int e = wv * 8 + (l15 >> 1);
            float m0  = fmaxf(s0, so0);
            float m1  = fmaxf(s1, so1);
            lds_logit[i + 8 * hlf][e]      = 10.0f * (m0 + __logf(__expf(s0 - m0) + __expf(so0 - m0)));
            lds_logit[16 + i + 8 * hlf][e] = 10.0f * (m1 + __logf(__expf(s1 - m1) + __expf(so1 - m1)));
        }
    }
    __syncthreads();

    // ---- epilogue 2: top-8 + masked softmax; wave handles 4 tokens ----
    const size_t TE = (size_t)TOKS * NEXP;
    #pragma unroll
    for (int tt = 0; tt < 4; ++tt) {
        const int t = wv * 4 + tt;
        float l0 = lds_logit[t][lane];
        float l1 = lds_logit[t][lane + 32];

        bool a0 = true, a1 = true;
        float thr = -1e30f, gmax = -1e30f;
        for (int it = 0; it < TOPK; ++it) {
            float cm = wred_max(fmaxf(a0 ? l0 : -1e30f, a1 ? l1 : -1e30f));
            int key = 0x7fffffff;
            if      (a0 && l0 == cm) key = lane * 2;
            else if (a1 && l1 == cm) key = lane * 2 + 1;
            key = wred_min_i(key);
            if (key == lane * 2)          a0 = false;
            else if (key == lane * 2 + 1) a1 = false;
            thr = cm;
            if (it == 0) gmax = cm;
        }
        const bool m0 = (l0 >= thr), m1 = (l1 >= thr);

        float e0 = __expf(l0 - gmax), e1 = __expf(l1 - gmax);
        float denom = wred_sum(e0 + e1);
        float d0 = e0 / denom, d1 = e1 / denom;
        float msum = wred_sum((m0 ? d0 : 0.f) + (m1 ? d1 : 0.f));
        float inv = 1.0f / (msum + 1e-9f);
        float p0 = m0 ? d0 * inv : 0.f;
        float p1 = m1 ? d1 * inv : 0.f;

        const size_t base = (size_t)(tokBase + t) * NEXP;
        out[base + lane]                = m0 ? 1.f : 0.f;   // mask_full
        out[base + lane + 32]           = m1 ? 1.f : 0.f;
        out[TE + base + lane]           = p0;               // probs
        out[TE + base + lane + 32]      = p1;
        out[2 * TE + base + lane]       = l0;               // logits_clean
        out[2 * TE + base + lane + 32]  = l1;
        out[3 * TE + base + lane]       = l0;               // logits_sel (temp = 1)
        out[3 * TE + base + lane + 32]  = l1;
    }
}

extern "C" void kernel_launch(void* const* d_in, const int* in_sizes, int n_in,
                              void* d_out, int out_size, void* d_ws, size_t ws_size,
                              hipStream_t stream) {
    const float* h      = (const float*)d_in[0];   // (T, D) fp32
    const float* protos = (const float*)d_in[1];   // (E, P, D) fp32

    // workspace: normalized bf16 prototypes (1 MB) + token inverse norms (64 KB)
    unsigned short* pb = (unsigned short*)d_ws;
    float* hn = (float*)((char*)d_ws + (size_t)NCOL * DIM * sizeof(unsigned short));
    float* out = (float*)d_out;

    hipLaunchKernelGGL(norm_cast_kernel, dim3(NCOL + TOKS), dim3(256), 0, stream,
                       h, protos, pb, hn);
    hipLaunchKernelGGL(router_kernel, dim3(TOKS / TILET), dim3(256), 0, stream,
                       h, pb, hn, out);
}
